// Attention_37074157699274
// MI455X (gfx1250) — compile-verified, matrix operand order checked by reading, not run on hardware
//
#include <hip/hip_runtime.h>
#include <hip/hip_bf16.h>

typedef __attribute__((ext_vector_type(16))) _Float16 v16h;
typedef __attribute__((ext_vector_type(8)))  _Float16 v8h;
typedef __attribute__((ext_vector_type(4)))  _Float16 v4h;
typedef __attribute__((ext_vector_type(8)))  float    v8f;
typedef __attribute__((ext_vector_type(4)))  float    v4f;

#define SEQ    32
#define DIM    256
#define INNER  512
#define NHEADS 8
#define DHEAD  64

// LDS strides in halfs (padded to dodge bank conflicts)
#define XS_STRIDE 264   // 256 + 8
#define O_STRIDE  520   // 512 + 8
#define QK_STRIDE 72    // 64 + 8
#define VT_STRIDE 40    // 32 + 8
#define AT_STRIDE 40    // 32 + 8

#define WAVE_SCR (SEQ*QK_STRIDE*2 + DHEAD*VT_STRIDE + SEQ*AT_STRIDE)  // 8448 halfs

#define WMMA_F16(a, b, c) \
  __builtin_amdgcn_wmma_f32_16x16x32_f16(false, (a), false, (b), (short)0, (c), false, false)

// Load one 16x32 f16 operand tile in the CDNA5 WMMA A/B register layout.
// Lane l (l<16): row = row0+l, K halfs {k0..k0+7, k0+16..k0+23}
// Lane l (l>=16): row = row0+l-16, K halfs {k0+8..k0+15, k0+24..k0+31}
// Requires rows contiguous in K with `stride` halfs between rows.
__device__ __forceinline__ v16h load_frag_h(const _Float16* base, int stride,
                                            int row0, int k0, int lane) {
  const int r  = row0 + (lane & 15);
  const int ks = k0 + ((lane >> 4) << 3);
  const _Float16* p = base + r * stride + ks;
  v8h lo = *(const v8h*)(p);
  v8h hi = *(const v8h*)(p + 16);
  v16h f;
#pragma unroll
  for (int i = 0; i < 8; ++i) { f[i] = lo[i]; f[i + 8] = hi[i]; }
  return f;
}

__device__ __forceinline__ float row_max16(float v) {
  v = fmaxf(v, __shfl_xor(v, 1));
  v = fmaxf(v, __shfl_xor(v, 2));
  v = fmaxf(v, __shfl_xor(v, 4));
  v = fmaxf(v, __shfl_xor(v, 8));
  return v;
}
__device__ __forceinline__ float row_sum16(float v) {
  v += __shfl_xor(v, 1);
  v += __shfl_xor(v, 2);
  v += __shfl_xor(v, 4);
  v += __shfl_xor(v, 8);
  return v;
}

// ---------------- Kernel 1: weight convert + transpose (f32 -> f16) ----------
// wqkvT: [1536][256]  row n holds W_qkv[:, n]   (K contiguous per row)
// woutT: [256][512]   row n holds W_out[:, n]   (K contiguous per row)
__global__ void convert_weights(const float* __restrict__ wqkv,
                                const float* __restrict__ wout,
                                _Float16* __restrict__ wqkvT,
                                _Float16* __restrict__ woutT) {
  const int i = blockIdx.x * blockDim.x + threadIdx.x;
  if (i < 1536 * 256) {
    const int n = i >> 8, k = i & 255;
    wqkvT[i] = (_Float16)wqkv[k * 1536 + n];
  }
  if (i < 256 * 512) {
    const int n = i >> 9, k = i & 511;
    woutT[i] = (_Float16)wout[k * 256 + n];
  }
}

// ---------------- Kernel 2: fused attention -------------------------------
__global__ void __launch_bounds__(128)
attn_kernel(const float* __restrict__ x,
            const _Float16* __restrict__ wqkvT,
            const _Float16* __restrict__ woutT,
            const float* __restrict__ bout,
            float* __restrict__ out) {
  __shared__ __align__(16) _Float16 xs[SEQ * XS_STRIDE];      // x tile, f16
  __shared__ __align__(16) _Float16 osm[SEQ * O_STRIDE];      // concat head outputs, f16
  __shared__ __align__(16) _Float16 wscr[4][WAVE_SCR];        // per-wave scratch

  const int tid  = threadIdx.x;
  const int wave = tid >> 5;
  const int lane = tid & 31;
  const int lhi  = lane >> 4;   // 0/1: row-half selector in C/D layout
  const int llo  = lane & 15;   // column within 16-wide tile
  const int bp   = blockIdx.x;

  const float* xt = x + (size_t)bp * (SEQ * DIM);

  // ---- stage x tile to LDS as f16 ----
#pragma unroll 4
  for (int i = tid; i < SEQ * (DIM / 4); i += 128) {
    const int m  = i >> 6;       // 0..31
    const int c4 = i & 63;       // float4 index within row
    v4f v = *(const v4f*)(xt + m * DIM + c4 * 4);
    v4h h;
    h[0] = (_Float16)v[0]; h[1] = (_Float16)v[1];
    h[2] = (_Float16)v[2]; h[3] = (_Float16)v[3];
    *(v4h*)(&xs[m * XS_STRIDE + c4 * 4]) = h;
  }
  __syncthreads();

  _Float16* qbuf = wscr[wave];
  _Float16* kbuf = qbuf + SEQ * QK_STRIDE;
  _Float16* vtb  = kbuf + SEQ * QK_STRIDE;   // V transposed: [64 d][32 j]
  _Float16* atb  = vtb + DHEAD * VT_STRIDE;  // attn probs:   [32 i][32 j]

  for (int hh = 0; hh < 2; ++hh) {
    const int head = wave * 2 + hh;

    // ---- QKV GEMMs: [32,256] x [256,64] per part ----
#pragma unroll
    for (int part = 0; part < 3; ++part) {
      const int ncol0 = part * INNER + head * DHEAD;  // column base in W_qkv
      v8f acc[2][4] = {};
#pragma unroll
      for (int kstep = 0; kstep < 8; ++kstep) {
        const int k0 = kstep * 32;
        v16h a0 = load_frag_h(xs, XS_STRIDE, 0,  k0, lane);
        v16h a1 = load_frag_h(xs, XS_STRIDE, 16, k0, lane);
#pragma unroll
        for (int nt = 0; nt < 4; ++nt) {
          v16h b = load_frag_h(wqkvT, DIM, ncol0 + nt * 16, k0, lane);
          acc[0][nt] = WMMA_F16(a0, b, acc[0][nt]);
          acc[1][nt] = WMMA_F16(a1, b, acc[1][nt]);
        }
      }
      if (part < 2) {
        // Q/K: store row-major [32][64] (rows contiguous in d)
        _Float16* dst = (part == 0) ? qbuf : kbuf;
#pragma unroll
        for (int mt = 0; mt < 2; ++mt)
#pragma unroll
          for (int nt = 0; nt < 4; ++nt)
#pragma unroll
            for (int r = 0; r < 8; ++r)
              dst[(mt * 16 + lhi * 8 + r) * QK_STRIDE + nt * 16 + llo] =
                  (_Float16)acc[mt][nt][r];
      } else {
        // V: store transposed [64 d][32 j] with vector stores
#pragma unroll
        for (int mt = 0; mt < 2; ++mt)
#pragma unroll
          for (int nt = 0; nt < 4; ++nt) {
            v8h p;
#pragma unroll
            for (int r = 0; r < 8; ++r) p[r] = (_Float16)acc[mt][nt][r];
            *(v8h*)(&vtb[(nt * 16 + llo) * VT_STRIDE + mt * 16 + lhi * 8]) = p;
          }
      }
    }
    __syncthreads();

    // ---- dots = Q K^T (K=64) ----
    v8f d[2][2] = {};
#pragma unroll
    for (int ks2 = 0; ks2 < 2; ++ks2) {
      const int k0 = ks2 * 32;
      v16h a0 = load_frag_h(qbuf, QK_STRIDE, 0,  k0, lane);
      v16h a1 = load_frag_h(qbuf, QK_STRIDE, 16, k0, lane);
      v16h b0 = load_frag_h(kbuf, QK_STRIDE, 0,  k0, lane);  // row j of K = col j of K^T
      v16h b1 = load_frag_h(kbuf, QK_STRIDE, 16, k0, lane);
      d[0][0] = WMMA_F16(a0, b0, d[0][0]);
      d[0][1] = WMMA_F16(a0, b1, d[0][1]);
      d[1][0] = WMMA_F16(a1, b0, d[1][0]);
      d[1][1] = WMMA_F16(a1, b1, d[1][1]);
    }

    // ---- scale + softmax over 32 columns (registers + shuffles) ----
    const float scale = 0.125f;  // 64^-0.5
#pragma unroll
    for (int mt = 0; mt < 2; ++mt) {
#pragma unroll
      for (int r = 0; r < 8; ++r) {
        const float va = d[mt][0][r] * scale;
        const float vb = d[mt][1][r] * scale;
        const float mx = row_max16(fmaxf(va, vb));
        const float ea = __expf(va - mx);
        const float eb = __expf(vb - mx);
        const float s  = row_sum16(ea + eb);
        const float inv = 1.0f / s;
        const int m = mt * 16 + lhi * 8 + r;
        atb[m * AT_STRIDE + llo]      = (_Float16)(ea * inv);
        atb[m * AT_STRIDE + 16 + llo] = (_Float16)(eb * inv);
      }
    }

    // ---- out_head = attn @ V  (K=32, one k-step) ----
    {
      v8f oa[2][4] = {};
      v16h a0 = load_frag_h(atb, AT_STRIDE, 0,  0, lane);
      v16h a1 = load_frag_h(atb, AT_STRIDE, 16, 0, lane);
#pragma unroll
      for (int nt = 0; nt < 4; ++nt) {
        v16h b = load_frag_h(vtb, VT_STRIDE, nt * 16, 0, lane);  // row d of Vt
        oa[0][nt] = WMMA_F16(a0, b, oa[0][nt]);
        oa[1][nt] = WMMA_F16(a1, b, oa[1][nt]);
      }
      // concat into shared [32][512]
#pragma unroll
      for (int mt = 0; mt < 2; ++mt)
#pragma unroll
        for (int nt = 0; nt < 4; ++nt)
#pragma unroll
          for (int r = 0; r < 8; ++r)
            osm[(mt * 16 + lhi * 8 + r) * O_STRIDE + head * DHEAD + nt * 16 + llo] =
                (_Float16)oa[mt][nt][r];
    }
    __syncthreads();
  }

  // ---- final projection: [32,512] @ [512,256] + bias ----
  const int ncol0 = wave * 64;  // each wave owns 64 output columns
  v8f acc[2][4] = {};
#pragma unroll
  for (int kstep = 0; kstep < 16; ++kstep) {
    const int k0 = kstep * 32;
    v16h a0 = load_frag_h(osm, O_STRIDE, 0,  k0, lane);
    v16h a1 = load_frag_h(osm, O_STRIDE, 16, k0, lane);
#pragma unroll
    for (int nt = 0; nt < 4; ++nt) {
      v16h b = load_frag_h(woutT, INNER, ncol0 + nt * 16, k0, lane);
      acc[0][nt] = WMMA_F16(a0, b, acc[0][nt]);
      acc[1][nt] = WMMA_F16(a1, b, acc[1][nt]);
    }
  }
  float* orow = out + (size_t)bp * (SEQ * DIM);
#pragma unroll
  for (int nt = 0; nt < 4; ++nt) {
    const int col = ncol0 + nt * 16 + llo;
    const float bv = bout[col];
#pragma unroll
    for (int mt = 0; mt < 2; ++mt)
#pragma unroll
      for (int r = 0; r < 8; ++r)
        orow[(mt * 16 + lhi * 8 + r) * DIM + col] = acc[mt][nt][r] + bv;
  }
}

extern "C" void kernel_launch(void* const* d_in, const int* in_sizes, int n_in,
                              void* d_out, int out_size, void* d_ws, size_t ws_size,
                              hipStream_t stream) {
  const float* x    = (const float*)d_in[0];   // [4,1024,32,256]
  const float* wqkv = (const float*)d_in[1];   // [256,1536]
  const float* wout = (const float*)d_in[2];   // [512,256]
  const float* bout = (const float*)d_in[3];   // [256]

  _Float16* wqkvT = (_Float16*)d_ws;            // [1536][256] f16
  _Float16* woutT = wqkvT + 1536 * 256;         // [256][512]  f16

  convert_weights<<<1536, 256, 0, stream>>>(wqkv, wout, wqkvT, woutT);

  const int n_bp = in_sizes[0] / (SEQ * DIM);   // 4096
  attn_kernel<<<n_bp, 128, 0, stream>>>(x, wqkvT, woutT, bout, (float*)d_out);
}
